// BLP_52467320487972
// MI455X (gfx1250) — compile-verified, matrix operand order checked by reading, not run on hardware
//
#include <hip/hip_runtime.h>
#include <hip/hip_bf16.h>

#define NUM_ENT   40000
#define EMBED     256
#define FEAT      768
#define KCHUNKS   (FEAT / 32)    // 24
#define NTILES    (EMBED / 16)   // 16
#define NE_TOT    40103          // 1 + NUM_ENT + 102
#define NE_SCORE  40001          // rows scored
#define BATCH     32

typedef __attribute__((ext_vector_type(16))) __bf16 v16bf;
typedef __attribute__((ext_vector_type(8)))  float  v8f;

// Per ISA layout (16-bit A/B fragment, 16x32 / 32x16): lane = row(A)/col(B)
// in [0,16), lane bit4 selects K-octet: half=0 holds K={0..7,16..23},
// half=1 holds K={8..15,24..31}; vector elems 0..7 -> first octet,
// elems 8..15 -> second octet (+16).
__device__ __forceinline__ void load16(const float* __restrict__ p, float* v) {
  float4 a0 = *(const float4*)(p + 0);
  float4 a1 = *(const float4*)(p + 4);
  float4 b0 = *(const float4*)(p + 16);
  float4 b1 = *(const float4*)(p + 20);
  v[0] = a0.x; v[1] = a0.y; v[2]  = a0.z; v[3]  = a0.w;
  v[4] = a1.x; v[5] = a1.y; v[6]  = a1.z; v[7]  = a1.w;
  v[8] = b0.x; v[9] = b0.y; v[10] = b0.z; v[11] = b0.w;
  v[12] = b1.x; v[13] = b1.y; v[14] = b1.z; v[15] = b1.w;
}

// One-shot: split proj_W (256x768 f32) into packed bf16 hi/lo B-fragments,
// stored in WMMA fragment order: [n_tile][k_chunk][lane][16 elems].
// One wave per (n_tile, k_chunk).
__global__ __launch_bounds__(32) void BLP_split_w(
    const float* __restrict__ W, __bf16* __restrict__ whi,
    __bf16* __restrict__ wlo) {
  const int lane = threadIdx.x;          // 0..31
  const int kc   = blockIdx.x % KCHUNKS; // 0..23
  const int nt   = blockIdx.x / KCHUNKS; // 0..15
  const int half = lane >> 4;
  const int r15  = lane & 15;

  const float* p = W + (size_t)(nt * 16 + r15) * FEAT + kc * 32 + half * 8;
  float v[16];
  load16(p, v);

  const size_t off = ((size_t)(nt * KCHUNKS + kc) * 32 + lane) * 16;
#pragma unroll
  for (int i = 0; i < 16; ++i) {
    __bf16 h = (__bf16)v[i];
    whi[off + i] = h;
    wlo[off + i] = (__bf16)(v[i] - (float)h);
  }
}

// ent_proj = ent_pkl (40000x768) @ proj_W^T, written into unity rows
// 1..40000 (row-major, stride 256).  Block = 128 threads (4 waves);
// wave computes 16 rows x 128 cols.  B fragments come pre-split/pre-packed
// from workspace (zero conversion VALU on the hot path).
__global__ __launch_bounds__(128) void BLP_gemm_proj(
    const float* __restrict__ A, const __bf16* __restrict__ whi,
    const __bf16* __restrict__ wlo, float* __restrict__ unity) {
  const int lane = threadIdx.x & 31;
  const int wave = threadIdx.x >> 5;
  const int half = lane >> 4;
  const int r15  = lane & 15;

  const int n_half = blockIdx.x & 1;                 // cols 0..127 / 128..255
  const int m_tile = (blockIdx.x >> 1) * 4 + wave;   // 0..2499
  const int m_base = m_tile * 16;

  v8f acc[8] = {};

  const float* arow = A + (size_t)(m_base + r15) * FEAT + half * 8;

  for (int kc = 0; kc < KCHUNKS; ++kc) {
    const int k0 = kc * 32;
    if (k0 + 32 < FEAT)
      __builtin_prefetch(arow + k0 + 32, 0, 0);   // global_prefetch next chunk

    // A fragment: load 16 f32, split to bf16 hi + residual lo
    float v[16];
    load16(arow + k0, v);
    v16bf ahi, alo;
#pragma unroll
    for (int i = 0; i < 16; ++i) {
      __bf16 h = (__bf16)v[i];
      ahi[i] = h;
      alo[i] = (__bf16)(v[i] - (float)h);
    }

#pragma unroll
    for (int nt = 0; nt < 8; ++nt) {
      const int nt_g = n_half * 8 + nt;
      const size_t off = ((size_t)(nt_g * KCHUNKS + kc) * 32 + lane) * 16;
      v16bf bhi = *(const v16bf*)(whi + off);      // 32B aligned vector load
      v16bf blo = *(const v16bf*)(wlo + off);
      // split-bf16 chain: hi*hi + hi*lo + lo*hi  (~f32 accuracy)
      acc[nt] = __builtin_amdgcn_wmma_f32_16x16x32_bf16(
          false, ahi, false, bhi, (short)0, acc[nt], false, false);
      acc[nt] = __builtin_amdgcn_wmma_f32_16x16x32_bf16(
          false, ahi, false, blo, (short)0, acc[nt], false, false);
      acc[nt] = __builtin_amdgcn_wmma_f32_16x16x32_bf16(
          false, alo, false, bhi, (short)0, acc[nt], false, false);
    }
  }

  // C/D layout: VGPR j, lanes 0-15 -> M=j, lanes 16-31 -> M=j+8; N = lane&15.
#pragma unroll
  for (int nt = 0; nt < 8; ++nt) {
#pragma unroll
    for (int j = 0; j < 8; ++j) {
      int m   = m_base + j + half * 8;
      int col = n_half * 128 + nt * 16 + r15;
      unity[(size_t)(m + 1) * EMBED + col] = acc[nt][j];
    }
  }
}

// unity[0] = other_emb[0];  unity[40001+i] = other_emb[1+i]
__global__ __launch_bounds__(256) void BLP_copy_other(
    const float* __restrict__ other, float* __restrict__ unity) {
  int r = blockIdx.x;                 // 0..102
  int d = threadIdx.x;                // 0..255
  int dst = (r == 0) ? 0 : (NUM_ENT + r);
  unity[(size_t)dst * EMBED + d] = other[(size_t)r * EMBED + d];
}

// One block per batch element: gather two rows, L2-normalize, sum.
__global__ __launch_bounds__(256) void BLP_query(
    const float* __restrict__ unity, const int* __restrict__ ids,
    const int* __restrict__ mp_ptr, float* __restrict__ qsum) {
  __shared__ float s0[256];
  __shared__ float s1[256];
  const int b = blockIdx.x;
  const int d = threadIdx.x;
  const int mp = *mp_ptr;
  const int c0 = (mp == 0) ? 1 : 0;          // kept cols = {0,1,2} \ {mp}
  const int c1 = (mp <= 1) ? 2 : 1;
  const int id0 = ids[b * 3 + c0];
  const int id1 = ids[b * 3 + c1];
  float v0 = unity[(size_t)id0 * EMBED + d];
  float v1 = unity[(size_t)id1 * EMBED + d];
  s0[d] = v0 * v0;
  s1[d] = v1 * v1;
  __syncthreads();
  for (int off = 128; off > 0; off >>= 1) {
    if (d < off) { s0[d] += s0[d + off]; s1[d] += s1[d + off]; }
    __syncthreads();
  }
  float r0 = 1.0f / fmaxf(sqrtf(s0[0]), 1e-12f);
  float r1 = 1.0f / fmaxf(sqrtf(s1[0]), 1e-12f);
  qsum[(size_t)b * EMBED + d] = v0 * r0 + v1 * r1;
}

// score[b][e] = -sum_d |qsum[b][d] - unity[e][d]|  for e in [0, 40001)
__global__ __launch_bounds__(256) void BLP_score(
    const float* __restrict__ unity, const float* __restrict__ qsum,
    float* __restrict__ out) {
  __shared__ float q[BATCH * EMBED];   // 32 KB, broadcast reads
  for (int i = threadIdx.x; i < BATCH * EMBED; i += 256) q[i] = qsum[i];
  __syncthreads();

  const int e = blockIdx.x * 256 + threadIdx.x;
  const bool active = (e < NE_SCORE);
  const float4* row = (const float4*)(unity + (size_t)(active ? e : 0) * EMBED);

  float acc[BATCH];
#pragma unroll
  for (int b = 0; b < BATCH; ++b) acc[b] = 0.0f;

  for (int d4 = 0; d4 < EMBED / 4; ++d4) {
    float4 v = row[d4];
    const int dbase = d4 * 4;
#pragma unroll
    for (int b = 0; b < BATCH; ++b) {
      const float* qb = q + b * EMBED + dbase;
      acc[b] += fabsf(qb[0] - v.x) + fabsf(qb[1] - v.y) +
                fabsf(qb[2] - v.z) + fabsf(qb[3] - v.w);
    }
  }
  if (active) {
#pragma unroll
    for (int b = 0; b < BATCH; ++b)
      out[(size_t)b * NE_SCORE + e] = -acc[b];   // coalesced per batch row
  }
}

extern "C" void kernel_launch(void* const* d_in, const int* in_sizes, int n_in,
                              void* d_out, int out_size, void* d_ws, size_t ws_size,
                              hipStream_t stream) {
  (void)in_sizes; (void)n_in; (void)out_size; (void)ws_size;
  const float* ent_pkl = (const float*)d_in[0];   // (40000, 768)
  const float* other   = (const float*)d_in[1];   // (103, 256)
  const float* proj_W  = (const float*)d_in[2];   // (256, 768)
  const int*   ids     = (const int*)d_in[3];     // (32, 3)
  const int*   mp      = (const int*)d_in[4];     // scalar
  float*       out     = (float*)d_out;           // (32, 40001)

  // Workspace layout (bytes): unity | qsum | whi | wlo
  float*  unity = (float*)d_ws;                              // 40103*256 f32
  float*  qsum  = unity + (size_t)NE_TOT * EMBED;            // 32*256 f32
  __bf16* whi   = (__bf16*)(qsum + (size_t)BATCH * EMBED);   // 32B-aligned
  __bf16* wlo   = whi + (size_t)NTILES * KCHUNKS * 32 * 16;  // +384KB

  BLP_split_w<<<NTILES * KCHUNKS, 32, 0, stream>>>(proj_W, whi, wlo);
  BLP_gemm_proj<<<1250, 128, 0, stream>>>(ent_pkl, whi, wlo, unity);
  BLP_copy_other<<<103, 256, 0, stream>>>(other, unity);
  BLP_query<<<BATCH, 256, 0, stream>>>(unity, ids, mp, qsum);
  BLP_score<<<(NE_SCORE + 255) / 256, 256, 0, stream>>>(unity, qsum, out);
}
